// QuantAttentionFused_72129680769340
// MI455X (gfx1250) — matrix operand order repaired and checked by reading, not compiled, not run on hardware
//
#include <hip/hip_runtime.h>
#include <hip/hip_bf16.h>
#include <stdint.h>

#define HIDDEN  4096
#define NHEADS  32
#define NKV     8
#define HDIM    128
#define SEQ     2048
#define QKV_OUT 6144   // (32 + 2*8) * 128

typedef __bf16 bf16;
typedef __attribute__((ext_vector_type(16))) __bf16 bf16x16;
typedef __attribute__((ext_vector_type(8)))  float  f32x8;

union Frag32B {
  bf16x16 v;
  uint4   q[2];
};

// A-fragment (16x32 bf16, MxK): lanes 0-15 row m, elems 0-15 = K0..15;
// lanes 16-31 same row, elems = K16..31.  row_k0 = &A[m][k0].
__device__ __forceinline__ bf16x16 load_a_frag(const bf16* row_k0, int hi) {
  Frag32B f;
  f.q[0] = *reinterpret_cast<const uint4*>(row_k0 + hi * 16);
  f.q[1] = *reinterpret_cast<const uint4*>(row_k0 + hi * 16 + 8);
  return f.v;
}

// B-fragment (32x16 bf16, KxN): lane holds column n = lane%16, 16 contiguous K
// elements starting at K-offset (lane/16)*16.  p = &Bt[n][k0 + (lane>>4)*16].
__device__ __forceinline__ bf16x16 load_b_frag(const bf16* p) {
  Frag32B f;
  f.q[0] = *reinterpret_cast<const uint4*>(p);
  f.q[1] = *reinterpret_cast<const uint4*>(p + 8);
  return f.v;
}

__device__ __forceinline__ f32x8 wmma_bf16(bf16x16 a, bf16x16 b, f32x8 c) {
  return __builtin_amdgcn_wmma_f32_16x16x32_bf16(false, a, false, b, (short)0, c,
                                                 false, false);
}

// xor-shuffle within wave32 via ds_bpermute (index bits [6:2] select lane)
__device__ __forceinline__ float lane_xor_f(float v, int mask, int lane) {
  return __int_as_float(
      __builtin_amdgcn_ds_bpermute(((lane ^ mask) << 2), __float_as_int(v)));
}

// async memory->LDS copy, 16B, tracked by ASYNCcnt.  lds_off = generic shared
// pointer truncated to 32 bits.
__device__ __forceinline__ void async_load_b128(unsigned lds_off,
                                                const bf16* gaddr) {
  asm volatile("global_load_async_to_lds_b128 %0, %1, off"
               :
               : "v"(lds_off), "v"(gaddr)
               : "memory");
}

// ---------------------------------------------------------------- casts
__global__ void cast_bf16_kernel(const float* __restrict__ in,
                                 bf16* __restrict__ out, int n) {
  int i = blockIdx.x * blockDim.x + threadIdx.x;
  if (i < n) out[i] = (bf16)in[i];
}

// in: [rows][cols] fp32 row-major  ->  out: [cols][rows] bf16 row-major
__global__ void transpose_cast_kernel(const float* __restrict__ in,
                                      bf16* __restrict__ out, int rows, int cols) {
  int i = blockIdx.x * blockDim.x + threadIdx.x;
  if (i >= rows * cols) return;
  int k = i % rows;
  int n = i / rows;
  out[i] = (bf16)in[(size_t)k * cols + n];
}

// ---------------------------------------------------------------- GEMM
// C[M][N] fp32 = A[M][K] bf16 (row-major) x Bt[N][K] bf16 (B transposed).
// 256 threads = 8 waves; block tile 128(M) x 128(N); wave w owns rows w*16..+16.
// B K-slice (128 x 32) staged once per workgroup into double-buffered LDS via
// GLOBAL_LOAD_ASYNC_TO_LDS_B128 (ASYNCcnt), shared by all 8 waves -> 8x less
// L2 traffic than register-direct B.  LDS rows padded to 40 elems (80B): the
// per-lane ds_load_b128 hits banks n*20 mod 64 -> conflict-free.
// All 16 ds_load_b128 are issued before the WMMA chain so the scheduler can
// use partial s_wait_dscnt and hide LDS latency under the matrix pipe.
__device__ __forceinline__ void gemm_step(const bf16 (*buf)[40],
                                          const bf16* arow, int k0, int lo,
                                          int hi, f32x8 acc[8]) {
  bf16x16 a = load_a_frag(arow + k0, hi);
  bf16x16 bfr[8];
#pragma unroll
  for (int j = 0; j < 8; ++j)
    bfr[j] = load_b_frag(&buf[j * 16 + lo][hi * 16]);
#pragma unroll
  for (int j = 0; j < 8; ++j) acc[j] = wmma_bf16(a, bfr[j], acc[j]);
}

__global__ void __launch_bounds__(256)
gemm_bf16_wmma(const bf16* __restrict__ A, const bf16* __restrict__ Bt,
               float* __restrict__ C, int M, int N, int K) {
  __shared__ __align__(16) bf16 Blds[2][128][40];
  const int t = threadIdx.x;
  const int lane = t & 31;
  const int w = t >> 5;                       // wave 0..7
  const int lo = lane & 15, hi = lane >> 4;
  const int nBase = blockIdx.x * 128;
  const int mBase = blockIdx.y * 128 + w * 16;

  // staging assignment: thread t moves 32B = row sn, k elems [sh, sh+16)
  const int sn = t >> 1;
  const int sh = (t & 1) * 16;
  const bf16* sprow = Bt + (size_t)(nBase + sn) * K + sh;
  const unsigned lds0 = (unsigned)(uintptr_t)&Blds[0][sn][sh];
  const unsigned lds1 = (unsigned)(uintptr_t)&Blds[1][sn][sh];

  // prologue: stage k0 = 0 into buffer 0
  async_load_b128(lds0, sprow);
  async_load_b128(lds0 + 16, sprow + 8);

  f32x8 acc[8] = {};
  const bf16* arow = A + (size_t)(mBase + lo) * K;
  const int KITER = K >> 5;

  // steady state: branch-free; last iteration peeled
  for (int it = 0; it < KITER - 1; ++it) {
    const int k0 = it << 5;
    // issue next K-slice into the other buffer, then wait only for the
    // current slice (the 2 just-issued loads stay outstanding)
    const unsigned lnext = (it & 1) ? lds0 : lds1;
    const bf16* g = sprow + k0 + 32;
    async_load_b128(lnext, g);
    async_load_b128(lnext + 16, g + 8);
    asm volatile("s_wait_asynccnt 0x2" ::: "memory");
    __syncthreads();
    gemm_step(Blds[it & 1], arow, k0, lo, hi, acc);
    __syncthreads();   // all waves done with this buffer before overwrite
  }
  // epilogue iteration
  asm volatile("s_wait_asynccnt 0x0" ::: "memory");
  __syncthreads();
  gemm_step(Blds[(KITER - 1) & 1], arow, (KITER - 1) << 5, lo, hi, acc);

  // D layout: elem i -> row mBase + hi*8 + i, col nBase + j*16 + lo
#pragma unroll
  for (int j = 0; j < 8; ++j)
#pragma unroll
    for (int i = 0; i < 8; ++i)
      C[(size_t)(mBase + hi * 8 + i) * N + nBase + j * 16 + lo] = acc[j][i];
}

// ---------------------------------------------------------------- RoPE + split
// xqkv fp32 [SEQ][6144]  ->  Qb bf16 [32][SEQ][128] (roped),
//                            Kb bf16 [8][SEQ][128] (roped),
//                            Vt bf16 [8][128][SEQ]  (transposed V)
__global__ void rope_split_kernel(const float* __restrict__ xqkv,
                                  bf16* __restrict__ Qb, bf16* __restrict__ Kb,
                                  bf16* __restrict__ Vt) {
  const int idx = blockIdx.x * blockDim.x + threadIdx.x;
  const int TOTAL_QK = (NHEADS + NKV) * SEQ * 64;
  if (idx < TOTAL_QK) {
    const int j = idx & 63;
    const int t = idx >> 6;
    const int s = t & (SEQ - 1);
    const int hh = t >> 11;                       // 0..39 : 32 Q heads + 8 K heads
    const float* row = xqkv + (size_t)s * QKV_OUT;
    // inv_freq = theta^(-j/64) = exp(-j * ln(10000)/64)
    const float ang = (float)s * __expf(-(float)j * 0.14391156f);
    float sn, cs;
    __sincosf(ang, &sn, &cs);
    const int off = (hh < NHEADS) ? hh * HDIM : HIDDEN + (hh - NHEADS) * HDIM;
    const float x1 = row[off + j];
    const float x2 = row[off + 64 + j];
    const float o1 = x1 * cs - x2 * sn;
    const float o2 = x1 * sn + x2 * cs;
    bf16* dst = (hh < NHEADS)
                    ? (Qb + ((size_t)hh * SEQ + s) * HDIM)
                    : (Kb + ((size_t)(hh - NHEADS) * SEQ + s) * HDIM);
    dst[j] = (bf16)o1;
    dst[64 + j] = (bf16)o2;
  } else {
    int v = idx - TOTAL_QK;
    if (v >= NKV * SEQ * HDIM) return;
    const int d = v & 127;
    const int t = v >> 7;
    const int s = t & (SEQ - 1);
    const int vh = t >> 11;
    Vt[((size_t)vh * HDIM + d) * SEQ + s] =
        (bf16)xqkv[(size_t)s * QKV_OUT + HIDDEN + NKV * HDIM + vh * HDIM + d];
  }
}

// ---------------------------------------------------------------- flash attention
// One wave: head h, 16 query rows. Online softmax over causal keys in 32-key steps.
__global__ void __launch_bounds__(32)
attn_kernel(const bf16* __restrict__ Qb, const bf16* __restrict__ Kb,
            const bf16* __restrict__ Vt, bf16* __restrict__ Ab) {
  __shared__ __align__(16) bf16 Plds[16][32];
  const int lane = threadIdx.x & 31;
  const int lo = lane & 15, hi = lane >> 4;
  const int qt = blockIdx.x;
  const int h = blockIdx.y;
  const int kvh = h >> 2;                        // N_KV_GROUPS = 4
  const int qBase = qt * 16;

  const bf16* Qh = Qb + (size_t)h * SEQ * HDIM;
  const bf16* Kh = Kb + (size_t)kvh * SEQ * HDIM;
  const bf16* Vh = Vt + (size_t)kvh * HDIM * SEQ;

  // Q tile 16x128: 4 A-fragments, loaded once and reused over all keys
  bf16x16 qf[4];
#pragma unroll
  for (int kk = 0; kk < 4; ++kk)
    qf[kk] = load_a_frag(Qh + (size_t)(qBase + lo) * HDIM + kk * 32, hi);

  f32x8 oacc[8] = {};
  float mrow[8], lrow[8];
#pragma unroll
  for (int i = 0; i < 8; ++i) { mrow[i] = -1e30f; lrow[i] = 0.0f; }

  const float scale = 0.08838834764831845f;      // 1/sqrt(128)

  for (int kb = 0; kb < qBase + 16; kb += 32) {
    // ---- scores: S[16 x 32] = Q x K^T  (row-major K == B-fragment layout)
    f32x8 s0 = {}, s1 = {};
#pragma unroll
    for (int kk = 0; kk < 4; ++kk) {
      const bf16* b0 = Kh + (size_t)(kb + lo) * HDIM + kk * 32 + hi * 16;
      const bf16* b1 = Kh + (size_t)(kb + 16 + lo) * HDIM + kk * 32 + hi * 16;
      s0 = wmma_bf16(qf[kk], load_b_frag(b0), s0);
      s1 = wmma_bf16(qf[kk], load_b_frag(b1), s1);
    }
    // ---- online softmax (per D-layout row: rows hi*8+i live on 16-lane group)
#pragma unroll
    for (int i = 0; i < 8; ++i) {
      const int q = qBase + hi * 8 + i;
      float v0 = s0[i] * scale;
      float v1 = s1[i] * scale;
      if (kb + lo > q) v0 = -1e30f;              // causal mask (== ref -1e9 mask)
      if (kb + 16 + lo > q) v1 = -1e30f;
      float rmax = fmaxf(v0, v1);
      rmax = fmaxf(rmax, lane_xor_f(rmax, 1, lane));
      rmax = fmaxf(rmax, lane_xor_f(rmax, 2, lane));
      rmax = fmaxf(rmax, lane_xor_f(rmax, 4, lane));
      rmax = fmaxf(rmax, lane_xor_f(rmax, 8, lane));
      const float mnew = fmaxf(mrow[i], rmax);
      const float alpha = __expf(mrow[i] - mnew);
      const float p0 = __expf(v0 - mnew);
      const float p1 = __expf(v1 - mnew);
      float rsum = p0 + p1;
      rsum += lane_xor_f(rsum, 1, lane);
      rsum += lane_xor_f(rsum, 2, lane);
      rsum += lane_xor_f(rsum, 4, lane);
      rsum += lane_xor_f(rsum, 8, lane);
      lrow[i] = lrow[i] * alpha + rsum;
      mrow[i] = mnew;
#pragma unroll
      for (int j = 0; j < 8; ++j) oacc[j][i] *= alpha;
      Plds[hi * 8 + i][lo] = (bf16)p0;
      Plds[hi * 8 + i][16 + lo] = (bf16)p1;
    }
    __syncthreads();
    // reload P in A-fragment layout (transpose via LDS)
    Frag32B pf;
    pf.q[0] = *reinterpret_cast<const uint4*>(&Plds[lo][hi * 16]);
    pf.q[1] = *reinterpret_cast<const uint4*>(&Plds[lo][hi * 16 + 8]);
    __syncthreads();
    // ---- O += P x V  (Vt dim-major -> contiguous B-fragments)
#pragma unroll
    for (int j = 0; j < 8; ++j) {
      const bf16* bv = Vh + (size_t)(j * 16 + lo) * SEQ + kb + hi * 16;
      oacc[j] = wmma_bf16(pf.v, load_b_frag(bv), oacc[j]);
    }
  }

  // normalize and store bf16 attn output [SEQ][HIDDEN], col = h*128 + j*16 + lo
#pragma unroll
  for (int i = 0; i < 8; ++i) {
    const float inv = 1.0f / lrow[i];
    const int q = qBase + hi * 8 + i;
#pragma unroll
    for (int j = 0; j < 8; ++j)
      Ab[(size_t)q * HIDDEN + h * HDIM + j * 16 + lo] = (bf16)(oacc[j][i] * inv);
  }
}

// ---------------------------------------------------------------- launch
extern "C" void kernel_launch(void* const* d_in, const int* in_sizes, int n_in,
                              void* d_out, int out_size, void* d_ws, size_t ws_size,
                              hipStream_t stream) {
  (void)in_sizes; (void)n_in; (void)out_size; (void)ws_size;
  const float* hidden = (const float*)d_in[0];
  const float* w_qkv  = (const float*)d_in[1];
  const float* w_o    = (const float*)d_in[2];
  // d_in[3] attention_mask is the causal mask; applied analytically in attn_kernel.
  float* out = (float*)d_out;

  char* ws = (char*)d_ws;
  // region map (bytes):
  //   [0,   16.8M)  Xb            -> later reused as Ab (same size)
  //   [16.8M, 67.1M) WqkvT        -> later reused as Qb/Kb/Vt
  //   [67.1M, 100.7M) WoT          (persistent)
  //   [100.7M, 151M)  xqkv fp32
  bf16*  Xb    = (bf16*)(ws + 0);
  bf16*  WqkvT = (bf16*)(ws + 16777216);
  bf16*  WoT   = (bf16*)(ws + 67108864);
  float* xqkv  = (float*)(ws + 100663296);
  bf16*  Qb    = (bf16*)(ws + 16777216);
  bf16*  Kb    = (bf16*)(ws + 33554432);
  bf16*  Vt    = (bf16*)(ws + 37748736);
  bf16*  Ab    = (bf16*)(ws + 0);

  cast_bf16_kernel<<<(SEQ * HIDDEN) / 256, 256, 0, stream>>>(hidden, Xb,
                                                             SEQ * HIDDEN);
  transpose_cast_kernel<<<(HIDDEN * QKV_OUT) / 256, 256, 0, stream>>>(
      w_qkv, WqkvT, HIDDEN, QKV_OUT);
  transpose_cast_kernel<<<(HIDDEN * HIDDEN) / 256, 256, 0, stream>>>(
      w_o, WoT, HIDDEN, HIDDEN);

  gemm_bf16_wmma<<<dim3(QKV_OUT / 128, SEQ / 128), 256, 0, stream>>>(
      Xb, WqkvT, xqkv, SEQ, QKV_OUT, HIDDEN);

  rope_split_kernel<<<((NHEADS + NKV) * SEQ * 64 + NKV * SEQ * HDIM) / 256, 256, 0,
                      stream>>>(xqkv, Qb, Kb, Vt);

  attn_kernel<<<dim3(SEQ / 16, NHEADS), 32, 0, stream>>>(Qb, Kb, Vt, Ab);

  gemm_bf16_wmma<<<dim3(HIDDEN / 128, SEQ / 128), 256, 0, stream>>>(
      Ab, WoT, out, SEQ, HIDDEN, HIDDEN);
}
